// Neck_63101659513255
// MI455X (gfx1250) — matrix operand order repaired and checked by reading, not compiled, not run on hardware
//
#include <hip/hip_runtime.h>
#include <hip/hip_bf16.h>

typedef __attribute__((ext_vector_type(16))) __bf16 bf16x16;
typedef __attribute__((ext_vector_type(8)))  float  floatx8;

static __device__ __forceinline__ unsigned short f2bf(float f) {
    unsigned int u = __float_as_uint(f);
    u += 0x7FFFu + ((u >> 16) & 1u);
    return (unsigned short)(u >> 16);
}

// ---------------------------------------------------------------------------
// 27-channel (18 offset + 9 modulation) 3x3 conv, stride 1, pad 1, fp32.
// ---------------------------------------------------------------------------
__global__ void offmod_conv(const float* __restrict__ X,
                            const float* __restrict__ w_off, const float* __restrict__ b_off,
                            const float* __restrict__ w_mod, const float* __restrict__ b_mod,
                            float* __restrict__ OFF, float* __restrict__ MODL,
                            int Cin, int H, int W) {
    int idx = blockIdx.x * blockDim.x + threadIdx.x;
    int HW = H * W;
    int x  = idx % W;
    int y  = (idx / W) % H;
    int ch = (idx / HW) % 27;
    int b  = idx / (27 * HW);
    const float* wsrc;
    float acc;
    if (ch < 18) { wsrc = w_off + (size_t)ch * Cin * 9;        acc = b_off[ch]; }
    else         { wsrc = w_mod + (size_t)(ch - 18) * Cin * 9; acc = b_mod[ch - 18]; }
    const float* xb = X + (size_t)b * Cin * HW;
    for (int c = 0; c < Cin; ++c) {
        const float* xc = xb + (size_t)c * HW;
        const float* wc = wsrc + c * 9;
        #pragma unroll
        for (int t = 0; t < 9; ++t) {
            int yy = y + t / 3 - 1;
            int xx = x + t % 3 - 1;
            if (yy >= 0 && yy < H && xx >= 0 && xx < W)
                acc += xc[yy * W + xx] * wc[t];
        }
    }
    if (ch < 18) OFF [((size_t)b * 18 + ch)        * HW + y * W + x] = acc;
    else         MODL[((size_t)b * 9  + (ch - 18)) * HW + y * W + x] = acc;
}

// ---------------------------------------------------------------------------
// Deformable bilinear sampling + modulation -> bf16 im2col A[M, Cin*9].
// ---------------------------------------------------------------------------
__global__ void deform_sample(const float* __restrict__ X,
                              const float* __restrict__ OFF, const float* __restrict__ MODL,
                              unsigned short* __restrict__ A,
                              int Cin, int H, int W) {
    int idx = blockIdx.x * blockDim.x + threadIdx.x;
    int HW = H * W;
    int K  = Cin * 9;
    int k  = idx % 9;
    int m  = idx / 9;            // m = b*HW + p
    int p  = m % HW;
    int b  = m / HW;
    int y  = p / W, x = p % W;

    float dy = OFF[((size_t)b * 18 + 2 * k)     * HW + p];
    float dx = OFF[((size_t)b * 18 + 2 * k + 1) * HW + p];
    float ml = MODL[((size_t)b * 9 + k) * HW + p];
    float mod = 1.0f / (1.0f + __expf(-ml));

    float py = dy + (float)(k / 3 - 1) + (float)y;
    float px = dx + (float)(k % 3 - 1) + (float)x;
    float y0f = floorf(py), x0f = floorf(px);
    float wy = py - y0f, wx = px - x0f;
    int y0 = (int)y0f, x0 = (int)x0f;
    int y1 = y0 + 1,   x1 = x0 + 1;
    float vy0 = (y0 >= 0 && y0 < H) ? 1.f : 0.f;
    float vy1 = (y1 >= 0 && y1 < H) ? 1.f : 0.f;
    float vx0 = (x0 >= 0 && x0 < W) ? 1.f : 0.f;
    float vx1 = (x1 >= 0 && x1 < W) ? 1.f : 0.f;
    int yc0 = min(max(y0, 0), H - 1), yc1 = min(max(y1, 0), H - 1);
    int xc0 = min(max(x0, 0), W - 1), xc1 = min(max(x1, 0), W - 1);
    float w00 = (1.f - wy) * (1.f - wx) * vy0 * vx0 * mod;
    float w01 = (1.f - wy) * wx         * vy0 * vx1 * mod;
    float w10 = wy         * (1.f - wx) * vy1 * vx0 * mod;
    float w11 = wy         * wx         * vy1 * vx1 * mod;
    int o00 = yc0 * W + xc0, o01 = yc0 * W + xc1;
    int o10 = yc1 * W + xc0, o11 = yc1 * W + xc1;

    const float* xb = X + (size_t)b * Cin * HW;
    size_t ab = (size_t)m * K + k;
    for (int c = 0; c < Cin; ++c) {
        const float* xc = xb + (size_t)c * HW;
        float g = xc[o00] * w00 + xc[o01] * w01 + xc[o10] * w10 + xc[o11] * w11;
        A[ab + (size_t)c * 9] = f2bf(g);
    }
}

// ---------------------------------------------------------------------------
// Elementwise fp32 -> bf16. Deform weights w[O][Cin*9] are already the
// transposed-B layout [N][K] the GEMM wants, so conversion is a pure cast.
// ---------------------------------------------------------------------------
__global__ void cvt_bf16(const float* __restrict__ src, unsigned short* __restrict__ dst) {
    int idx = blockIdx.x * blockDim.x + threadIdx.x;
    dst[idx] = f2bf(src[idx]);
}

// ---------------------------------------------------------------------------
// Transposed-conv weights per output parity, stored transposed [N][K]:
// wt[Cin][Cout][3][3] fp32 -> Wt[par][Cout][Cin*4] bf16.
// parity py: even -> ky=1 only; odd -> ky in {0,2}. Same for x.
// ---------------------------------------------------------------------------
__global__ void prep_w_tconv(const float* __restrict__ wt, unsigned short* __restrict__ Wt,
                             int Cin, int Cout) {
    int idx = blockIdx.x * blockDim.x + threadIdx.x;
    int K2  = Cin * 4;
    int k2  = idx % K2;
    int co  = (idx / K2) % Cout;
    int par = idx / (K2 * Cout);
    int t = k2 & 3, ci = k2 >> 2;
    int py = par >> 1, px = par & 1;
    int tyi = t >> 1,  txi = t & 1;
    int nty = py ? 2 : 1, ntx = px ? 2 : 1;
    float v = 0.f;
    if (tyi < nty && txi < ntx) {
        int ky = py ? (tyi ? 2 : 0) : 1;
        int kx = px ? (txi ? 2 : 0) : 1;
        v = wt[((size_t)ci * Cout + co) * 9 + ky * 3 + kx];
    }
    Wt[((size_t)par * Cout + co) * K2 + k2] = f2bf(v);
}

// ---------------------------------------------------------------------------
// im2col for transposed conv: Y[B,Cin,H,W] fp32 -> A[par][B*H*W][Cin*4] bf16.
// ---------------------------------------------------------------------------
__global__ void tconv_im2col(const float* __restrict__ Y, unsigned short* __restrict__ A,
                             int Cin, int H, int W) {
    int idx = blockIdx.x * blockDim.x + threadIdx.x;
    int HW = H * W;
    int K  = Cin * 4;
    int p   = idx % HW;
    int b   = (idx / HW) % 8;
    int par = idx / (8 * HW);
    int y = p / W, x = p % W;
    int py = par >> 1, px = par & 1;

    int   iy[2]; float vy[2];
    int   ix[2]; float vx[2];
    if (py == 0) { iy[0] = y; vy[0] = 1.f; iy[1] = 0; vy[1] = 0.f; }
    else {
        vy[0] = (y + 1 < H) ? 1.f : 0.f;
        iy[0] = (y + 1 < H) ? (y + 1) : (H - 1);
        iy[1] = y; vy[1] = 1.f;
    }
    if (px == 0) { ix[0] = x; vx[0] = 1.f; ix[1] = 0; vx[1] = 0.f; }
    else {
        vx[0] = (x + 1 < W) ? 1.f : 0.f;
        ix[0] = (x + 1 < W) ? (x + 1) : (W - 1);
        ix[1] = x; vx[1] = 1.f;
    }
    float wv[4]; int off[4];
    #pragma unroll
    for (int t = 0; t < 4; ++t) {
        int tyi = t >> 1, txi = t & 1;
        wv[t]  = vy[tyi] * vx[txi];
        off[t] = iy[tyi] * W + ix[txi];
    }
    int m = b * HW + p;
    size_t Mp = (size_t)8 * HW;
    size_t ab = ((size_t)par * Mp + m) * K;
    const float* yb = Y + (size_t)b * Cin * HW;
    for (int c = 0; c < Cin; ++c) {
        const float* yc = yb + (size_t)c * HW;
        ushort4 pk;
        pk.x = f2bf(yc[off[0]] * wv[0]);
        pk.y = f2bf(yc[off[1]] * wv[1]);
        pk.z = f2bf(yc[off[2]] * wv[2]);
        pk.w = f2bf(yc[off[3]] * wv[3]);
        *(ushort4*)(A + ab + (size_t)c * 4) = pk;
    }
}

// ---------------------------------------------------------------------------
// WMMA bf16 GEMM: Out = A[M,K] * Bt[N,K]^T + bias[N], fp32 accumulate.
// Block = 256 threads (8 waves), tile 128x64x32; each wave -> 32(M)x32(N)
// = 4 accumulators -> 4 v_wmma per K-step. Tiles staged to LDS with
// gfx1250 async loads (ASYNCcnt) so the VALU stays free for WMMA.
// Output scattered to NCHW with (oy0,ox0,step) pixel mapping.
// ---------------------------------------------------------------------------
__global__ void __launch_bounds__(256)
gemm_wmma_bf16(const unsigned short* __restrict__ A, const unsigned short* __restrict__ Bt,
               const float* __restrict__ bias, float* __restrict__ Out,
               int M, int N, int K,
               int W, int HW, int OW, int OHW, int oy0, int ox0, int step) {
    __shared__ __align__(16) unsigned short sA[128 * 32];   // [row][k]   8 KB
    __shared__ __align__(16) unsigned short sB[64 * 32];    // [col][k]   4 KB

    int tid  = threadIdx.x;
    int lane = tid & 31;
    int wid  = tid >> 5;
    int wave_m = wid >> 1;          // 0..3 -> M offset wave_m*32
    int wave_n = wid & 1;           // 0..1 -> N offset wave_n*32
    int n0 = blockIdx.x * 64;
    int m0 = blockIdx.y * 128;

    floatx8 acc[2][2] = {};

    // Cooperative staging assignments (all contiguous b128 moves):
    int arow = tid >> 1, ahalf = (tid & 1) * 16;   // A: 128 rows x 2 halves of 16 elems
    int brow = tid >> 2, bchunk = (tid & 3) * 8;   // B: 64 rows x 4 chunks of 8 elems
    const unsigned short* aptr = A  + (size_t)(m0 + arow) * K + ahalf;
    const unsigned short* bptr = Bt + (size_t)(n0 + brow) * K + bchunk;
    unsigned ldsa0 = (unsigned)(uintptr_t)(sA + arow * 32 + ahalf);
    unsigned ldsa1 = (unsigned)(uintptr_t)(sA + arow * 32 + ahalf + 8);
    unsigned ldsb  = (unsigned)(uintptr_t)(sB + brow * 32 + bchunk);

    for (int k0 = 0; k0 < K; k0 += 32) {
        // gfx1250 async global->LDS staging (tracked by ASYNCcnt)
        asm volatile("global_load_async_to_lds_b128 %0, %1, off"
                     :: "v"(ldsa0), "v"(aptr + k0) : "memory");
        asm volatile("global_load_async_to_lds_b128 %0, %1, off"
                     :: "v"(ldsa1), "v"(aptr + k0 + 8) : "memory");
        asm volatile("global_load_async_to_lds_b128 %0, %1, off"
                     :: "v"(ldsb), "v"(bptr + k0) : "memory");
        if (k0 + 32 < K) {
            __builtin_prefetch(aptr + k0 + 32, 0, 0);
            __builtin_prefetch(bptr + k0 + 32, 0, 0);
        }
        asm volatile("s_wait_asynccnt 0x0" ::: "memory");
        __syncthreads();

        // Fragments per CDNA5 wave32 layouts.
        int khalf = lane >> 4, l15 = lane & 15;
        union U { uint4 q[2]; bf16x16 v; };
        U af[2], bf[2];
        #pragma unroll
        for (int mi = 0; mi < 2; ++mi) {
            int row = wave_m * 32 + mi * 16 + l15;
            af[mi].q[0] = *(const uint4*)(sA + row * 32 + khalf * 8);        // K khalf*8..+7
            af[mi].q[1] = *(const uint4*)(sA + row * 32 + 16 + khalf * 8);   // K 16+khalf*8..+7
        }
        #pragma unroll
        for (int ni = 0; ni < 2; ++ni) {
            int col = wave_n * 32 + ni * 16 + l15;
            bf[ni].q[0] = *(const uint4*)(sB + col * 32 + khalf * 8);        // K khalf*8..+7
            bf[ni].q[1] = *(const uint4*)(sB + col * 32 + 16 + khalf * 8);   // K 16+khalf*8..+7
        }
        #pragma unroll
        for (int mi = 0; mi < 2; ++mi)
            #pragma unroll
            for (int ni = 0; ni < 2; ++ni)
                acc[mi][ni] = __builtin_amdgcn_wmma_f32_16x16x32_bf16(
                    false, af[mi].v, false, bf[ni].v, (short)0, acc[mi][ni], false, false);
        __syncthreads();
    }

    // Epilogue: D layout -> row = j + 8*(lane>>4), col = lane&15; NCHW scatter.
    int khalf = lane >> 4, l15 = lane & 15;
    #pragma unroll
    for (int ni = 0; ni < 2; ++ni) {
        int coln = n0 + wave_n * 32 + ni * 16 + l15;
        float bia = bias[coln];
        #pragma unroll
        for (int mi = 0; mi < 2; ++mi) {
            #pragma unroll
            for (int j = 0; j < 8; ++j) {
                int m = m0 + wave_m * 32 + mi * 16 + khalf * 8 + j;
                int b = m / HW;
                int r = m % HW;
                int y = r / W, x = r % W;
                int oy = oy0 + y * step, ox = ox0 + x * step;
                size_t ob = (size_t)b * N * OHW + (size_t)oy * OW + ox;
                Out[ob + (size_t)coln * OHW] = acc[mi][ni][j] + bia;
            }
        }
    }
}

// ---------------------------------------------------------------------------
// Host orchestration.
// ---------------------------------------------------------------------------
extern "C" void kernel_launch(void* const* d_in, const int* in_sizes, int n_in,
                              void* d_out, int out_size, void* d_ws, size_t ws_size,
                              hipStream_t stream) {
    (void)in_sizes; (void)n_in; (void)out_size; (void)ws_size;
    const int B = 8;
    const int CinA[3]  = {512, 256, 128};
    const int CoutA[3] = {256, 128, 64};
    const int HA[3]    = {32, 64, 128};

    char*  ws  = (char*)d_ws;
    size_t off = 0;
    auto alloc = [&](size_t bytes) -> char* {
        char* p = ws + off;
        off += (bytes + 255) & ~(size_t)255;
        return p;
    };
    // Max across stages: A im2col 150,994,944 bf16 elems (stage3 deform).
    unsigned short* Abuf  = (unsigned short*)alloc(302000128ULL);
    float*          Ybuf  = (float*)alloc(33554432ULL);   // deform output, max stage3
    float*          X2    = (float*)alloc(33554432ULL);   // [8,256,64,64]
    float*          X3    = (float*)alloc(67108864ULL);   // [8,128,128,128]
    float*          OFFb  = (float*)alloc(9437184ULL);    // [8,18,128,128]
    float*          MODLb = (float*)alloc(4718592ULL);    // [8,9,128,128]
    unsigned short* Wm    = (unsigned short*)alloc(2359296ULL);  // [N][K] bf16
    unsigned short* Wt    = (unsigned short*)alloc(2097152ULL);  // [4][N][K] bf16

    const float* Xs = (const float*)d_in[0];
    for (int s = 0; s < 3; ++s) {
        int Cin = CinA[s], Cout = CoutA[s], H = HA[s], Wd = HA[s];
        int HW = H * Wd;
        const float* w_off = (const float*)d_in[1 + s * 8 + 0];
        const float* b_off = (const float*)d_in[1 + s * 8 + 1];
        const float* w_mod = (const float*)d_in[1 + s * 8 + 2];
        const float* b_mod = (const float*)d_in[1 + s * 8 + 3];
        const float* w     = (const float*)d_in[1 + s * 8 + 4];
        const float* bw    = (const float*)d_in[1 + s * 8 + 5];
        const float* wt    = (const float*)d_in[1 + s * 8 + 6];
        const float* bt    = (const float*)d_in[1 + s * 8 + 7];
        float* Xn = (s == 0) ? X2 : (s == 1) ? X3 : (float*)d_out;

        // 1) offset + modulation conv (27 ch, fp32 direct)
        offmod_conv<<<(B * 27 * HW) / 256, 256, 0, stream>>>(
            Xs, w_off, b_off, w_mod, b_mod, OFFb, MODLb, Cin, H, Wd);

        // 2) deform weights: w[O][Cin*9] is already the [N][K] layout -> cast
        cvt_bf16<<<(Cout * Cin * 9) / 256, 256, 0, stream>>>(w, Wm);

        // 3) deformable sampling -> bf16 im2col
        deform_sample<<<(B * HW * 9) / 256, 256, 0, stream>>>(
            Xs, OFFb, MODLb, Abuf, Cin, H, Wd);

        // 4) deform GEMM -> Y (NCHW fp32)
        int M = B * HW, N = Cout, K = Cin * 9;
        dim3 g1(N / 64, M / 128);
        gemm_wmma_bf16<<<g1, 256, 0, stream>>>(Abuf, Wm, bw, Ybuf,
                                               M, N, K, Wd, HW, Wd, HW, 0, 0, 1);

        // 5) transposed conv: parity im2col + weights + 4 GEMMs -> Xn (2H x 2W)
        tconv_im2col<<<(4 * B * HW) / 256, 256, 0, stream>>>(Ybuf, Abuf, Cout, H, Wd);
        prep_w_tconv<<<(16 * Cout * Cout) / 256, 256, 0, stream>>>(wt, Wt, Cout, Cout);
        int K2 = Cout * 4;
        size_t Mp = (size_t)B * HW;
        for (int par = 0; par < 4; ++par) {
            dim3 g2(Cout / 64, M / 128);
            gemm_wmma_bf16<<<g2, 256, 0, stream>>>(
                Abuf + (size_t)par * Mp * K2,
                Wt + ((size_t)par * Cout) * K2,
                bt, Xn, M, Cout, K2,
                Wd, HW, 2 * Wd, 4 * HW, par >> 1, par & 1, 2);
        }
        Xs = Xn;
    }
}
